// SelfAttention_11123965296874
// MI455X (gfx1250) — matrix operand order repaired and checked by reading, not compiled, not run on hardware
//
#include <hip/hip_runtime.h>
#include <hip/hip_bf16.h>
#include <stdint.h>

// Self-attention for b=2, L=2048, E=1024, heads=16, dh=64 on gfx1250 (MI455X).
// Matrix work on v_wmma_f32_16x16x32_bf16 (wave32, fp32 accum); GEMM and K/V tile staging
// via global_load_async_to_lds_b128 (ASYNCcnt) with double buffering so the DMA
// overlaps the WMMA stream.

#define SEQ    2048
#define EMBED  1024
#define HEADS  16
#define DHEAD  64
#define NBATCH 2

typedef __attribute__((ext_vector_type(16))) __bf16 v16bf;
typedef __attribute__((ext_vector_type(8)))  float  v8f;

union Frag16 { unsigned int u[8]; v16bf v; };
union AccF   { float f[8]; v8f v; };

__device__ __forceinline__ unsigned short f2bf(float f) {
  unsigned int x = __float_as_uint(f);
  x += 0x7FFFu + ((x >> 16) & 1u);       // round-to-nearest-even
  return (unsigned short)(x >> 16);
}

// Low 32 bits of a generic pointer into LDS == byte offset in the wave's LDS
// allocation (flat LDS aperture keeps the offset in addr[31:0]).
__device__ __forceinline__ unsigned ldsoff(const void* p) {
  return (unsigned)(uintptr_t)p;
}

// CDNA5 async DMA: copy 16B/lane from global to LDS, tracked by ASYNCcnt.
__device__ __forceinline__ void gl_async_b128(unsigned lds_addr, const void* gptr) {
  asm volatile("global_load_async_to_lds_b128 %0, %1, off"
               :: "v"(lds_addr), "v"(gptr) : "memory");
}
__device__ __forceinline__ void wait_async0() {
  asm volatile("s_wait_asynccnt 0" ::: "memory");
}

// 16-bit A matrix 16x32 (MxK) fragment from LDS, row stride `ld` (multiple of 8).
// Lane L: M = L&15; elements j=0..7 -> K = j + 8*(L>>4); j=8..15 -> K = j+8+8*(L>>4).
__device__ __forceinline__ void load_frag_a(const unsigned short* base, int ld, Frag16& fr) {
  const int lane = threadIdx.x & 31;
  const unsigned short* row = base + (lane & 15) * ld + 8 * (lane >> 4);
  const uint4 lo = *(const uint4*)(row);        // K = 0..7   (+8*half)
  const uint4 hi = *(const uint4*)(row + 16);   // K = 16..23 (+8*half)
  fr.u[0] = lo.x; fr.u[1] = lo.y; fr.u[2] = lo.z; fr.u[3] = lo.w;
  fr.u[4] = hi.x; fr.u[5] = hi.y; fr.u[6] = hi.z; fr.u[7] = hi.w;
}

// 16-bit B matrix 32x16 (KxN): lane L holds column N = L&15; elements j=0..7 -> K = j + 8*(L>>4);
// j=8..15 -> K = j+8+8*(L>>4). `base` points at row 0 of the Bt tile (rows=N, contiguous K).
__device__ __forceinline__ void load_frag_b(const unsigned short* base, int ld, Frag16& fr) {
  const int lane = threadIdx.x & 31;
  const unsigned short* row = base + (lane & 15) * ld + 8 * (lane >> 4);
  const uint4 lo = *(const uint4*)(row);        // K = 0..7   (+8*half)
  const uint4 hi = *(const uint4*)(row + 16);   // K = 16..23 (+8*half)
  fr.u[0] = lo.x; fr.u[1] = lo.y; fr.u[2] = lo.z; fr.u[3] = lo.w;
  fr.u[4] = hi.x; fr.u[5] = hi.y; fr.u[6] = hi.z; fr.u[7] = hi.w;
}

__device__ __forceinline__ v8f wmma_bf16(const Frag16& a, const Frag16& b, v8f c) {
  return __builtin_amdgcn_wmma_f32_16x16x32_bf16(
      /*neg_a=*/false, a.v, /*neg_b=*/false, b.v,
      /*c_mod=*/(short)0, c, /*reuse_a=*/false, /*reuse_b=*/false);
}

// ---------------------------------------------------------------- cvt fp32->bf16
__global__ __launch_bounds__(256) void cvt_f32_bf16(const float* __restrict__ in,
                                                    unsigned short* __restrict__ out,
                                                    int n) {
  int i = (blockIdx.x * 256 + threadIdx.x) * 4;
  if (i + 3 < n) {
    float4 f = *(const float4*)(in + i);
    uint2 p;
    p.x = (unsigned int)f2bf(f.x) | ((unsigned int)f2bf(f.y) << 16);
    p.y = (unsigned int)f2bf(f.z) | ((unsigned int)f2bf(f.w) << 16);
    *(uint2*)(out + i) = p;
  }
}

// ----------------------------------------------------- generic C = A * B^T GEMM
// A: [M,K] bf16 row-major; B: [N,K] bf16 row-major; C: [M,N] fp32.
// Block tile 64x128, 256 threads (8 waves): wave w -> rows (w>>1)*16, cols (w&1)*64.
// Double-buffered async-DMA K slices of 32.
#define GLDT 40   // LDS row stride (ushorts) for a 32-wide K slice; 80B, 16B-aligned
__global__ __launch_bounds__(256) void gemm_bf16_nt(const unsigned short* __restrict__ A,
                                                    const unsigned short* __restrict__ B,
                                                    float* __restrict__ C,
                                                    int M, int N, int K) {
  __shared__ unsigned short As[2 * 64 * GLDT];
  __shared__ unsigned short Bs[2 * 128 * GLDT];
  const int t    = threadIdx.x;
  const int wave = t >> 5;
  const int mi   = wave >> 1, ni = wave & 1;
  const int row0 = blockIdx.x * 64, col0 = blockIdx.y * 128;

  AccF acc[4];
#pragma unroll
  for (int c = 0; c < 4; ++c)
#pragma unroll
    for (int r = 0; r < 8; ++r) acc[c].f[r] = 0.f;

  const int ar = t >> 2, aq = t & 3;   // A tile: 64 rows x 4 x 16B chunks
  auto issue_tiles = [&](int k0, int buf) {
    gl_async_b128(ldsoff(As + buf * 64 * GLDT + ar * GLDT + aq * 8),
                  A + (size_t)(row0 + ar) * K + k0 + aq * 8);
#pragma unroll
    for (int i = 0; i < 2; ++i) {
      const int c = t + i * 256;        // B tile: 128 rows x 4 x 16B chunks
      const int br = c >> 2, bq = c & 3;
      gl_async_b128(ldsoff(Bs + buf * 128 * GLDT + br * GLDT + bq * 8),
                    B + (size_t)(col0 + br) * K + k0 + bq * 8);
    }
  };

  issue_tiles(0, 0);
  wait_async0();
  __syncthreads();

  const int nsteps = K >> 5;
  for (int s = 0; s < nsteps; ++s) {
    const int buf = s & 1;
    if (s + 1 < nsteps) issue_tiles((s + 1) * 32, buf ^ 1);  // DMA overlaps WMMA
    Frag16 fa;
    load_frag_a(As + buf * 64 * GLDT + mi * 16 * GLDT, GLDT, fa);
#pragma unroll
    for (int c = 0; c < 4; ++c) {
      Frag16 fb;
      load_frag_b(Bs + buf * 128 * GLDT + (ni * 64 + c * 16) * GLDT, GLDT, fb);
      acc[c].v = wmma_bf16(fa, fb, acc[c].v);
    }
    wait_async0();
    __syncthreads();
  }

  // C/D layout: VGPR r -> row r + 8*(lane>=16), col = lane&15 within the n-tile.
  const int lane = t & 31, nn = lane & 15, hf = lane >> 4;
#pragma unroll
  for (int c = 0; c < 4; ++c) {
    float* cp = C + (size_t)(row0 + mi * 16 + 8 * hf) * N + col0 + ni * 64 + c * 16 + nn;
#pragma unroll
    for (int r = 0; r < 8; ++r) cp[(size_t)r * N] = acc[c].f[r];
  }
}

// ------------------------------------------- per-head L2-norm + RoPE + layouts
// y: [b, l, 3*1024] fp32 (head h owns cols [192h,192h+192): q|k|v, 64 each).
// Outputs: qbf/kbf [b,h,l,64] bf16 ; vt [b,h,64,l] bf16 (V transposed for PV GEMM).
// One wave per row; lane i handles the RoPE pair (i, i+32).
__global__ __launch_bounds__(256) void normrope(const float* __restrict__ y,
                                                const float* __restrict__ scale_p,
                                                unsigned short* __restrict__ qbf,
                                                unsigned short* __restrict__ kbf,
                                                unsigned short* __restrict__ vt) {
  const int bid   = blockIdx.x;          // 512 blocks: b*256 + h*16 + chunk
  const int chunk = bid & 15;
  const int h     = (bid >> 4) & 15;
  const int b     = bid >> 8;
  const int lane  = threadIdx.x & 31;
  const int wave  = threadIdx.x >> 5;
  const float es   = __expf(scale_p[0]);
  const float invf = __powf(10000.f, -(float)lane / 32.f);  // inv_freq[lane]
  const size_t bh  = (size_t)(b * HEADS + h);

  for (int li = 0; li < 16; ++li) {
    const int l = chunk * 128 + li * 8 + wave;
    const float* yr = y + (size_t)(b * SEQ + l) * (3 * EMBED) + h * (3 * DHEAD);
    float qa = yr[lane],       qb = yr[lane + 32];
    float ka = yr[64 + lane],  kb = yr[96 + lane];
    float va = yr[128 + lane], vb = yr[160 + lane];
    float sq = qa * qa + qb * qb, sk = ka * ka + kb * kb;
#pragma unroll
    for (int sh = 16; sh; sh >>= 1) {
      sq += __shfl_xor(sq, sh, 32);
      sk += __shfl_xor(sk, sh, 32);
    }
    const float qs = es * rsqrtf(fmaxf(sq, 1e-24f));  // /max(||q||,1e-12)*e^scale
    const float ks = rsqrtf(fmaxf(sk, 1e-24f));
    qa *= qs; qb *= qs; ka *= ks; kb *= ks;
    float sn, cs;
    __sincosf((float)l * invf, &sn, &cs);
    const float q0 = qa * cs - qb * sn, q1 = qb * cs + qa * sn;  // rotate-half
    const float k0 = ka * cs - kb * sn, k1 = kb * cs + ka * sn;
    const size_t qo = (bh * SEQ + l) * DHEAD;
    qbf[qo + lane]      = f2bf(q0);
    qbf[qo + lane + 32] = f2bf(q1);
    kbf[qo + lane]      = f2bf(k0);
    kbf[qo + lane + 32] = f2bf(k1);
    vt[(bh * DHEAD + lane)      * SEQ + l] = f2bf(va);
    vt[(bh * DHEAD + lane + 32) * SEQ + l] = f2bf(vb);
  }
}

// -------------------------------------------------------------- flash attention
// One block per (b, h, 64-row q tile); 128 threads = 4 waves, each owns 16 rows.
// K/V 64-key tiles double-buffered via async DMA. Per-row softmax stats live
// redundantly across the 16-lane half-group matching the WMMA C layout
// (row = r + 8*(lane>=16)); reductions via shfl_xor {1,2,4,8}.
#define FLD 72  // LDS row stride (ushorts): 144B, 16B-aligned, de-conflicted
__global__ __launch_bounds__(128) void flash_attn(const unsigned short* __restrict__ qbf,
                                                  const unsigned short* __restrict__ kbf,
                                                  const unsigned short* __restrict__ vt,
                                                  unsigned short* __restrict__ obf) {
  __shared__ unsigned short Qs[64 * FLD];
  __shared__ unsigned short Ks[2 * 64 * FLD];     // 64 keys x 64 d, double buffered
  __shared__ unsigned short Vs[2 * 64 * FLD];     // 64 d x 64 keys (V^T), dbl buf
  __shared__ unsigned short Ps[4][16 * FLD];      // per-wave P re-staging

  const int bid  = blockIdx.x;                    // b*512 + h*32 + qt
  const int qt   = bid & 31;
  const int h    = (bid >> 5) & 15;
  const int b    = bid >> 9;
  const size_t bh = (size_t)(b * HEADS + h);
  const int t = threadIdx.x, wave = t >> 5, lane = t & 31;
  const int nn = lane & 15, hf = lane >> 4;

  const unsigned short* qp = qbf + (bh * SEQ + qt * 64) * DHEAD;
  const unsigned short* kp = kbf + bh * SEQ * DHEAD;
  const unsigned short* vp = vt + bh * DHEAD * SEQ;

  auto issue_kv = [&](int kt, int buf) {
#pragma unroll
    for (int i = 0; i < 4; ++i) {
      const int c = t + i * 128;        // 64 rows x 8 x 16B chunks each
      const int r = c >> 3, q = c & 7;
      gl_async_b128(ldsoff(Ks + buf * 64 * FLD + r * FLD + q * 8),
                    kp + (size_t)kt * 64 * DHEAD + r * DHEAD + q * 8);
      gl_async_b128(ldsoff(Vs + buf * 64 * FLD + r * FLD + q * 8),
                    vp + (size_t)r * SEQ + kt * 64 + q * 8);
    }
  };

  // stage Q tile (64x64) once + first K/V tile
  issue_kv(0, 0);
#pragma unroll
  for (int i = 0; i < 4; ++i) {
    const int c = t + i * 128;
    const int r = c >> 3, q = c & 7;
    *(uint4*)(Qs + r * FLD + q * 8) = *(const uint4*)(qp + r * DHEAD + q * 8);
  }

  float rm[8], rl[8];
  AccF o[4];
#pragma unroll
  for (int r = 0; r < 8; ++r) { rm[r] = -3.0e38f; rl[r] = 0.f; }
#pragma unroll
  for (int c = 0; c < 4; ++c)
#pragma unroll
    for (int r = 0; r < 8; ++r) o[c].f[r] = 0.f;

  wait_async0();
  __syncthreads();

  for (int kt = 0; kt < SEQ / 64; ++kt) {
    const int buf = kt & 1;
    if (kt + 1 < SEQ / 64) issue_kv(kt + 1, buf ^ 1);   // DMA overlaps compute
    const unsigned short* ksb = Ks + buf * 64 * FLD;
    const unsigned short* vsb = Vs + buf * 64 * FLD;

    // S(16x64) = Q_wave * K^T
    AccF s[4];
#pragma unroll
    for (int c = 0; c < 4; ++c)
#pragma unroll
      for (int r = 0; r < 8; ++r) s[c].f[r] = 0.f;
#pragma unroll
    for (int ksw = 0; ksw < 2; ++ksw) {
      Frag16 fa;
      load_frag_a(Qs + wave * 16 * FLD + ksw * 32, FLD, fa);
#pragma unroll
      for (int c = 0; c < 4; ++c) {
        Frag16 fb;
        load_frag_b(ksb + c * 16 * FLD + ksw * 32, FLD, fb);
        s[c].v = wmma_bf16(fa, fb, s[c].v);
      }
    }

    // online softmax
    float mx[8], al[8], ts[8];
#pragma unroll
    for (int r = 0; r < 8; ++r) {
      float v = fmaxf(fmaxf(s[0].f[r], s[1].f[r]), fmaxf(s[2].f[r], s[3].f[r]));
#pragma unroll
      for (int sh = 1; sh < 16; sh <<= 1) v = fmaxf(v, __shfl_xor(v, sh, 32));
      mx[r] = fmaxf(v, rm[r]);
      al[r] = __expf(rm[r] - mx[r]);
      ts[r] = 0.f;
    }
    unsigned short* pw = Ps[wave];
#pragma unroll
    for (int c = 0; c < 4; ++c)
#pragma unroll
      for (int r = 0; r < 8; ++r) {
        const float p = __expf(s[c].f[r] - mx[r]);
        ts[r] += p;
        pw[(r + 8 * hf) * FLD + c * 16 + nn] = f2bf(p);  // C-layout -> A-layout
      }
#pragma unroll
    for (int r = 0; r < 8; ++r) {
      float v = ts[r];
#pragma unroll
      for (int sh = 1; sh < 16; sh <<= 1) v += __shfl_xor(v, sh, 32);
      rl[r] = rl[r] * al[r] + v;
      rm[r] = mx[r];
    }
#pragma unroll
    for (int c = 0; c < 4; ++c)
#pragma unroll
      for (int r = 0; r < 8; ++r) o[c].f[r] *= al[r];

    // O += P(16x64) * V(64x64); B rows = d dims (V^T tile), K = keys.
#pragma unroll
    for (int ksw = 0; ksw < 2; ++ksw) {
      Frag16 fa;
      load_frag_a(pw + ksw * 32, FLD, fa);   // same-wave LDS RAW: DS ops in-order
#pragma unroll
      for (int c = 0; c < 4; ++c) {
        Frag16 fb;
        load_frag_b(vsb + c * 16 * FLD + ksw * 32, FLD, fb);
        o[c].v = wmma_bf16(fa, fb, o[c].v);
      }
    }
    wait_async0();     // next tile's DMA done; barrier also fences readers
    __syncthreads();
  }

  // finalize: O /= rowsum, write bf16 [b, l, h*64+d] for the output GEMM
#pragma unroll
  for (int r = 0; r < 8; ++r) rl[r] = 1.f / fmaxf(rl[r], 1e-30f);
  const int row0 = qt * 64 + wave * 16 + 8 * hf;
#pragma unroll
  for (int c = 0; c < 4; ++c)
#pragma unroll
    for (int r = 0; r < 8; ++r) {
      const size_t off =
          ((size_t)(b * SEQ) + row0 + r) * EMBED + h * DHEAD + c * 16 + nn;
      obf[off] = f2bf(o[c].f[r] * rl[r]);
    }
}

// -------------------------------------------------------------------- launcher
extern "C" void kernel_launch(void* const* d_in, const int* in_sizes, int n_in,
                              void* d_out, int out_size, void* d_ws, size_t ws_size,
                              hipStream_t stream) {
  const float* x     = (const float*)d_in[0];   // [2,2048,1024]
  const float* Wqkv  = (const float*)d_in[1];   // [3072,1024]
  const float* Wout  = (const float*)d_in[2];   // [1024,1024]
  const float* scale = (const float*)d_in[3];   // scalar
  float* out = (float*)d_out;                   // [2,2048,1024]

  char* ws = (char*)d_ws;                       // 96 MB used
  const size_t MB = (size_t)1 << 20;
  unsigned short* xbf    = (unsigned short*)(ws + 0 * MB);   //  8 MB
  unsigned short* wqkvbf = (unsigned short*)(ws + 8 * MB);   //  6 MB
  unsigned short* woutbf = (unsigned short*)(ws + 14 * MB);  //  2 MB
  float*          y      = (float*)         (ws + 16 * MB);  // 48 MB [4096,3072]
  unsigned short* qbf    = (unsigned short*)(ws + 64 * MB);  //  8 MB
  unsigned short* kbf    = (unsigned short*)(ws + 72 * MB);  //  8 MB
  unsigned short* vt     = (unsigned short*)(ws + 80 * MB);  //  8 MB
  unsigned short* obf    = (unsigned short*)(ws + 88 * MB);  //  8 MB

  const int M = NBATCH * SEQ;                   // 4096

  // 1) fp32 -> bf16 operand conversion
  cvt_f32_bf16<<<(M * EMBED / 4 + 255) / 256, 256, 0, stream>>>(x, xbf, M * EMBED);
  cvt_f32_bf16<<<(3 * EMBED * EMBED / 4 + 255) / 256, 256, 0, stream>>>(
      Wqkv, wqkvbf, 3 * EMBED * EMBED);
  cvt_f32_bf16<<<(EMBED * EMBED / 4 + 255) / 256, 256, 0, stream>>>(
      Wout, woutbf, EMBED * EMBED);

  // 2) y = x @ W_qkv^T : [4096,1024] x [3072,1024]^T -> [4096,3072]
  gemm_bf16_nt<<<dim3(M / 64, (3 * EMBED) / 128), 256, 0, stream>>>(
      xbf, wqkvbf, y, M, 3 * EMBED, EMBED);

  // 3) head-dim norm + exp(scale) + RoPE; emit bf16 Q,K and V^T
  normrope<<<NBATCH * HEADS * 16, 256, 0, stream>>>(y, scale, qbf, kbf, vt);

  // 4) flash attention per (b, h, 64-row q tile)
  flash_attn<<<NBATCH * HEADS * (SEQ / 64), 128, 0, stream>>>(qbf, kbf, vt, obf);

  // 5) out = O @ W_out^T : [4096,1024] x [1024,1024]^T -> fp32 d_out
  gemm_bf16_nt<<<dim3(M / 64, EMBED / 128), 256, 0, stream>>>(
      obf, woutbf, out, M, EMBED, EMBED);
}
